// CrossAttention_44375602102869
// MI455X (gfx1250) — matrix operand order repaired and checked by reading, not compiled, not run on hardware
//
#include <hip/hip_runtime.h>
#include <hip/hip_bf16.h>

typedef __bf16 bf16_t;
typedef __attribute__((ext_vector_type(16))) __bf16 v16bf;
typedef __attribute__((ext_vector_type(8)))  __bf16 v8bf;
typedef __attribute__((ext_vector_type(8)))  float  v8f;
typedef __attribute__((ext_vector_type(4)))  int    v4i;

#define DIM      1024
#define HEADS    16
#define HEAD_DIM 64
#define SEQ      2048
#define BATCH    2
#define ROWS     (BATCH * SEQ)      // 4096 flattened rows
#define ATT_SCALE 0.125f            // 64^-0.5

#define WMMA_BF16(a, b, c) \
  __builtin_amdgcn_wmma_f32_16x16x32_bf16(false, (a), false, (b), (short)0, (c), false, false)

// Fragment assembly without 16-bit element moves
union U16bf {
  v16bf v;
  v8bf  h[2];
  unsigned u[8];
};

static __device__ __forceinline__ bf16_t f2bf(float f) {
  unsigned u = __builtin_bit_cast(unsigned, f);
  unsigned r = u + 0x7FFFu + ((u >> 16) & 1u);   // round-to-nearest-even
  unsigned short h = (unsigned short)(r >> 16);
  return __builtin_bit_cast(bf16_t, h);
}

// ---------------------------------------------------------------------------
// gfx1250 async global->LDS staging (ASYNCcnt path), with safe fallback
// ---------------------------------------------------------------------------
#if defined(__has_builtin)
#  if __has_builtin(__builtin_amdgcn_global_load_async_to_lds_b128)
#    define HAVE_ASYNC_LDS 1
#  endif
#endif
#ifndef HAVE_ASYNC_LDS
#  define HAVE_ASYNC_LDS 0
#endif

static __device__ __forceinline__ void stage16(const bf16_t* g, bf16_t* l) {
#if HAVE_ASYNC_LDS
  __builtin_amdgcn_global_load_async_to_lds_b128(
      (__attribute__((address_space(1))) v4i*)g,
      (__attribute__((address_space(3))) v4i*)l, 0, 0);
#else
  *(v8bf*)l = *(const v8bf*)g;    // 16-byte copy through VGPRs
#endif
}
static __device__ __forceinline__ void stage_join() {
#if HAVE_ASYNC_LDS
  asm volatile("s_wait_asynccnt 0" ::: "memory");
#endif
}

// ---------------------------------------------------------------------------
// Elementwise f32 -> bf16 convert
// ---------------------------------------------------------------------------
__global__ void k_f32_to_bf16(const float* __restrict__ in, bf16_t* __restrict__ out, int n) {
  int i = blockIdx.x * blockDim.x + threadIdx.x;
  if (i < n) out[i] = f2bf(in[i]);
}

// W [K][N] f32 -> WT [N][K] bf16 (so WMMA B-fragments are K-contiguous)
__global__ void k_transpose_bf16(const float* __restrict__ W, bf16_t* __restrict__ WT) {
  int i = blockIdx.x * blockDim.x + threadIdx.x;   // over DIM*DIM
  int k = i >> 10;
  int n = i & (DIM - 1);
  WT[(size_t)n * DIM + k] = f2bf(W[(size_t)k * DIM + n]);
}

// ---------------------------------------------------------------------------
// bf16 WMMA GEMM: C[R,N] = A[R,K] * W[K,N]  (W given transposed: WT[N][K])
// Block = 256 threads = 8 waves; wave tile 16x64; block tile 128x64.
// B tile (identical across waves) staged through LDS each k-step.
// ---------------------------------------------------------------------------
template <bool BF16OUT>
__global__ void __launch_bounds__(256) k_gemm_bf16(const bf16_t* __restrict__ A,
                                                   const bf16_t* __restrict__ WT,
                                                   const float* __restrict__ bias,
                                                   void* __restrict__ out,
                                                   int R, int N, int K) {
  __shared__ bf16_t Bl[64][48];   // 64 cols x 32 k (stride 48 elems = 96B, 32B-aligned rows)

  const int tid  = threadIdx.x;
  const int lane = tid & 31;
  const int wave = tid >> 5;
  const int hw   = lane >> 4;
  const int l16  = lane & 15;
  const int row0 = blockIdx.x * 128 + wave * 16;
  const int col0 = blockIdx.y * 64;
  const int scol = tid >> 2;        // 0..63   staging: column
  const int sk   = (tid & 3) * 8;   // 0,8,16,24  staging: k offset

  v8f acc[4] = {};
  const bf16_t* arow = A + (size_t)(row0 + l16) * K;
  const bf16_t* brow = WT + (size_t)(col0 + scol) * K + sk;

  for (int k = 0; k < K; k += 32) {
    __syncthreads();                              // previous tile fully consumed
    stage16(brow + k, &Bl[scol][sk]);
    stage_join();
    __syncthreads();

    // A-fragment 16x32: lane row = l16; K(e) = 16*hw + e contiguous
    U16bf a;
    a.h[0] = *(const v8bf*)(arow + k + 16 * hw);
    a.h[1] = *(const v8bf*)(arow + k + 16 * hw + 8);
#pragma unroll
    for (int c = 0; c < 4; ++c) {
      // B-fragment 32x16: lane col = l16; k = 16*hw + e contiguous
      v16bf b = *(const v16bf*)&Bl[c * 16 + l16][16 * hw];
      acc[c] = WMMA_BF16(a.v, b, acc[c]);
    }
  }

#pragma unroll
  for (int c = 0; c < 4; ++c) {
#pragma unroll
    for (int r = 0; r < 8; ++r) {
      int row = row0 + r + 8 * hw;   // C/D layout: VGPR r -> row r + 8*half
      int col = col0 + c * 16 + l16;
      if constexpr (BF16OUT) {
        ((bf16_t*)out)[(size_t)row * N + col] = f2bf(acc[c][r]);
      } else {
        ((float*)out)[(size_t)row * N + col] = acc[c][r] + bias[col];
      }
    }
  }
}

// ---------------------------------------------------------------------------
// Flash attention: block = 8 waves = 128 queries for one (b,h).
// K and V (transposed) tiles staged cooperatively in LDS, 32 keys per step.
// Online softmax; P transposed via wave-local LDS tile.
// ---------------------------------------------------------------------------
__global__ void __launch_bounds__(256) k_flash_attn(const bf16_t* __restrict__ Qb,
                                                    const bf16_t* __restrict__ Kb,
                                                    const bf16_t* __restrict__ Vb,
                                                    bf16_t* __restrict__ Ob) {
  __shared__ bf16_t Kl[32][80];      // 32 keys x 64 d   (stride 80 = 160B, 32B-aligned)
  __shared__ bf16_t Vt[64][48];      // V transposed: [d][key] (stride 48 = 96B)
  __shared__ bf16_t Pl[8][16][34];   // per-wave 16x32 P tile (stride 34 = 68B, 4B-aligned)

  const int tid  = threadIdx.x;
  const int lane = tid & 31;
  const int wave = tid >> 5;
  const int hw   = lane >> 4;
  const int l16  = lane & 15;
  const int b    = blockIdx.z;
  const int h    = blockIdx.y;
  const int q0   = blockIdx.x * 128 + wave * 16;

  const size_t base = (size_t)b * SEQ * DIM + (size_t)h * HEAD_DIM;
  const bf16_t* Q = Qb + base;
  const bf16_t* K = Kb + base;
  const bf16_t* V = Vb + base;

  // staging maps
  const int skey = tid >> 3;          // 0..31: K tile row
  const int sd   = (tid & 7) * 8;     // 0..56: K tile d offset (16B chunks)
  const int vkey = tid & 31;          // V tile key
  const int vd   = (tid >> 5) * 8;    // V tile d offset

  // Q as two persistent A-fragments (d 0..31 and 32..63)
  U16bf aq[2];
#pragma unroll
  for (int f = 0; f < 2; ++f) {
    const bf16_t* qr = Q + (size_t)(q0 + l16) * DIM + f * 32;
    aq[f].h[0] = *(const v8bf*)(qr + 16 * hw);
    aq[f].h[1] = *(const v8bf*)(qr + 16 * hw + 8);
  }

  v8f oA[4] = {};
  float m8[8], l8[8];
#pragma unroll
  for (int r = 0; r < 8; ++r) { m8[r] = -1e30f; l8[r] = 0.0f; }

  for (int j = 0; j < SEQ; j += 32) {
    // ---- cooperative staging of K (row-major) and V (transposed) ----
    __syncthreads();                       // previous tiles fully consumed
    stage16(K + (size_t)(j + skey) * DIM + sd, &Kl[skey][sd]);
    {
      v8bf vv = *(const v8bf*)(V + (size_t)(j + vkey) * DIM + vd);
#pragma unroll
      for (int e = 0; e < 8; ++e) Vt[vd + e][vkey] = vv[e];
    }
    stage_join();
    __syncthreads();

    // ---- S = Q*K^T for 32 keys (two 16x16 D tiles) ----
    v8f s0 = {}, s1 = {};
    {
      v16bf b0 = *(const v16bf*)&Kl[l16][16 * hw];
      v16bf b1 = *(const v16bf*)&Kl[l16][32 + 16 * hw];
      s0 = WMMA_BF16(aq[0].v, b0, s0);
      s0 = WMMA_BF16(aq[1].v, b1, s0);
      v16bf b2 = *(const v16bf*)&Kl[16 + l16][16 * hw];
      v16bf b3 = *(const v16bf*)&Kl[16 + l16][32 + 16 * hw];
      s1 = WMMA_BF16(aq[0].v, b2, s1);
      s1 = WMMA_BF16(aq[1].v, b3, s1);
    }

    // ---- online softmax (rows live in one 16-lane half; xor masks 1..8) ----
    float p0[8], p1[8], fac[8];
#pragma unroll
    for (int r = 0; r < 8; ++r) {
      float a = s0[r] * ATT_SCALE;
      float c = s1[r] * ATT_SCALE;
      float t = fmaxf(a, c);
      t = fmaxf(t, __shfl_xor(t, 1, 32));
      t = fmaxf(t, __shfl_xor(t, 2, 32));
      t = fmaxf(t, __shfl_xor(t, 4, 32));
      t = fmaxf(t, __shfl_xor(t, 8, 32));
      float nm = fmaxf(m8[r], t);
      fac[r] = __expf(m8[r] - nm);
      p0[r]  = __expf(a - nm);
      p1[r]  = __expf(c - nm);
      float rs = p0[r] + p1[r];
      rs += __shfl_xor(rs, 1, 32);
      rs += __shfl_xor(rs, 2, 32);
      rs += __shfl_xor(rs, 4, 32);
      rs += __shfl_xor(rs, 8, 32);
      l8[r] = l8[r] * fac[r] + rs;
      m8[r] = nm;
      oA[0][r] *= fac[r]; oA[1][r] *= fac[r]; oA[2][r] *= fac[r]; oA[3][r] *= fac[r];
    }

    // ---- transpose P (C-layout -> A-layout) through wave-local LDS ----
#pragma unroll
    for (int r = 0; r < 8; ++r) {
      Pl[wave][r + 8 * hw][l16]      = f2bf(p0[r]);
      Pl[wave][r + 8 * hw][16 + l16] = f2bf(p1[r]);
    }
    asm volatile("s_wait_dscnt 0" ::: "memory");
    U16bf pa;                          // read as 32-bit pairs: k = 2i(+1) + 16*hw (+8)
#pragma unroll
    for (int i = 0; i < 4; ++i) {
      pa.u[i]     = *(const unsigned*)&Pl[wave][l16][2 * i + 16 * hw];
      pa.u[4 + i] = *(const unsigned*)&Pl[wave][l16][8 + 2 * i + 16 * hw];
    }

    // ---- O += P * V (V B-fragments contiguous from transposed LDS tile) ----
#pragma unroll
    for (int c = 0; c < 4; ++c) {
      v16bf vb = *(const v16bf*)&Vt[c * 16 + l16][16 * hw];
      oA[c] = WMMA_BF16(pa.v, vb, oA[c]);
    }
  }

  // ---- normalize and store O (bf16, [b, n, h*64+d] layout) ----
#pragma unroll
  for (int r = 0; r < 8; ++r) {
    float inv = 1.0f / l8[r];
    size_t orow = base + (size_t)(q0 + r + 8 * hw) * DIM;
#pragma unroll
    for (int c = 0; c < 4; ++c)
      Ob[orow + c * 16 + l16] = f2bf(oA[c][r] * inv);
  }
}

// ---------------------------------------------------------------------------
extern "C" void kernel_launch(void* const* d_in, const int* in_sizes, int n_in,
                              void* d_out, int out_size, void* d_ws, size_t ws_size,
                              hipStream_t stream) {
  (void)in_sizes; (void)n_in; (void)out_size; (void)ws_size;

  const float* x  = (const float*)d_in[0];
  const float* y  = (const float*)d_in[1];
  const float* Wq = (const float*)d_in[2];
  const float* Wk = (const float*)d_in[3];
  const float* Wv = (const float*)d_in[4];
  const float* Wp = (const float*)d_in[5];
  const float* bp = (const float*)d_in[6];
  float* out = (float*)d_out;

  char* ws = (char*)d_ws;
  size_t off = 0;
  auto carve = [&](size_t bytes) {
    char* p = ws + off;
    off += (bytes + 255) & ~(size_t)255;
    return p;
  };
  const size_t XY_B = (size_t)ROWS * DIM * sizeof(bf16_t);   // 8 MB
  const size_t W_B  = (size_t)DIM * DIM * sizeof(bf16_t);    // 2 MB
  bf16_t* xb  = (bf16_t*)carve(XY_B);
  bf16_t* yb  = (bf16_t*)carve(XY_B);
  bf16_t* WqT = (bf16_t*)carve(W_B);
  bf16_t* WkT = (bf16_t*)carve(W_B);
  bf16_t* WvT = (bf16_t*)carve(W_B);
  bf16_t* WpT = (bf16_t*)carve(W_B);
  bf16_t* Qb  = (bf16_t*)carve(XY_B);
  bf16_t* Kbf = (bf16_t*)carve(XY_B);
  bf16_t* Vbf = (bf16_t*)carve(XY_B);
  bf16_t* Ob  = (bf16_t*)carve(XY_B);

  const int n_xy = ROWS * DIM;                 // 4,194,304
  k_f32_to_bf16<<<n_xy / 256, 256, 0, stream>>>(x, xb, n_xy);
  k_f32_to_bf16<<<n_xy / 256, 256, 0, stream>>>(y, yb, n_xy);

  const int n_w = DIM * DIM;                   // 1,048,576
  k_transpose_bf16<<<n_w / 256, 256, 0, stream>>>(Wq, WqT);
  k_transpose_bf16<<<n_w / 256, 256, 0, stream>>>(Wk, WkT);
  k_transpose_bf16<<<n_w / 256, 256, 0, stream>>>(Wv, WvT);
  k_transpose_bf16<<<n_w / 256, 256, 0, stream>>>(Wp, WpT);

  dim3 gg(ROWS / 128, DIM / 64);               // (32, 16)
  k_gemm_bf16<true><<<gg, 256, 0, stream>>>(xb, WqT, nullptr, Qb,  ROWS, DIM, DIM);
  k_gemm_bf16<true><<<gg, 256, 0, stream>>>(yb, WkT, nullptr, Kbf, ROWS, DIM, DIM);
  k_gemm_bf16<true><<<gg, 256, 0, stream>>>(yb, WvT, nullptr, Vbf, ROWS, DIM, DIM);

  dim3 ga(SEQ / 128, HEADS, BATCH);            // (16, 16, 2)
  k_flash_attn<<<ga, 256, 0, stream>>>(Qb, Kbf, Vbf, Ob);

  k_gemm_bf16<false><<<gg, 256, 0, stream>>>(Ob, WpT, bp, out, ROWS, DIM, DIM);
}
